// MetaFormerBlock_89876485636340
// MI455X (gfx1250) — matrix operand order repaired and checked by reading, not compiled, not run on hardware
//
#include <hip/hip_runtime.h>

// ---------------------------------------------------------------------------
// MetaFormer block for MI455X (gfx1250). All matrix math (pointwise convs,
// MLP, block-diagonal spectral matmuls, AND the 2D FFT/iFFT expressed as
// twiddle-matrix GEMMs) runs through v_wmma_f32_16x16x32_bf16 with f32
// accumulation. bf16 inputs keep fp32 exponent range (activations are
// LayerNorm'd, twiddles are O(1)); K=32 per WMMA issue is 8x the f32 path.
// Round-2 changes: B staged transposed in LDS so B fragments load as
// ds_load_b128 (like A); A tiles staged via global_load_async_to_lds_b128
// (ASYNCcnt) on the stride-1-K GEMMs.
// ---------------------------------------------------------------------------

typedef __attribute__((ext_vector_type(16))) __bf16 v16bf;
typedef __attribute__((ext_vector_type(8)))  float  v8f;

#define MT 128   // M tile
#define NT 64    // N tile
#define KT 32    // K step (one WMMA K)
#define AP 40    // LDS A row pitch (halves): 80B, 16B-aligned rows, 20 banks
#define KP 40    // LDS B row pitch (halves), B stored [n][k]

struct GemmP {
  const __bf16* A;  const __bf16* A2;           // dual-input: acc = A*B + A2*B2
  long sAm, sAk, sAb1, sAb2;                    // A strides (elems): m, k, batch%div, batch/div
  const __bf16* B;  const __bf16* B2;
  long sBk, sBn, sBb1, sBb2;
  float* Cf; __bf16* Cb;                        // either/both outputs
  long sCm, sCn, sCb1, sCb2;
  const float*  bias;  long sBias;              // bias[z*sBias + n]
  const __bf16* mult;                           // elementwise *= (C indexing)
  const float*  addres;                         // elementwise += (C indexing)
  const float*  star_s; const float* star_b;    // StarReLU scalars (device ptrs)
  int M, N, K, batchDiv, use_star;
};

__global__ __launch_bounds__(256) void kgemm(GemmP p) {
  __shared__ __bf16 As [MT*AP];
  __shared__ __bf16 As2[MT*AP];
  __shared__ __bf16 Bs [NT*KP];
  __shared__ __bf16 Bs2[NT*KP];

  const int tid  = threadIdx.x;
  const int lane = tid & 31;
  const int wave = tid >> 5;
  const int wm   = wave & 3;           // 4x2 wave grid over the 128x64 tile
  const int wn   = wave >> 2;
  const int r    = lane & 15;
  const int hi   = lane >> 4;

  const int z   = blockIdx.z;
  const int zb2 = z / p.batchDiv;
  const int zb1 = z - zb2 * p.batchDiv;
  const long offA = (long)zb2 * p.sAb2 + (long)zb1 * p.sAb1;
  const long offB = (long)zb2 * p.sBb2 + (long)zb1 * p.sBb1;
  const long offC = (long)zb2 * p.sCb2 + (long)zb1 * p.sCb1;
  const int m0 = blockIdx.x * MT;
  const int n0 = blockIdx.y * NT;
  const bool dual = (p.A2 != nullptr);

  // A fast path: stride-1 K, full tile, 16B-aligned rows -> async DMA to LDS
  const bool fastA_base = (p.sAk == 1) && ((p.sAm & 7) == 0) && ((offA & 7) == 0)
                          && (m0 + MT <= p.M);

  v8f acc[2][2];
  {
    v8f zz = {0.f,0.f,0.f,0.f,0.f,0.f,0.f,0.f};
    acc[0][0]=zz; acc[0][1]=zz; acc[1][0]=zz; acc[1][1]=zz;
  }

  for (int k0 = 0; k0 < p.K; k0 += KT) {
    const bool fastA = fastA_base && (k0 + KT <= p.K);
    if (fastA) {
      // ---- async DMA: 128x32 halves = 8KB, 2 x b128 per thread -----------
      #pragma unroll
      for (int i = 0; i < 2; ++i) {
        int chunk = tid + i*256;                 // 0..511
        int m = chunk >> 2;
        int k = (chunk & 3) * 8;
        long ge = offA + (long)(m0+m)*p.sAm + (long)(k0+k);
        unsigned ldst = (unsigned)(unsigned long long)(As + (m*AP + k));
        asm volatile("global_load_async_to_lds_b128 %0, %1, off"
                     :: "v"(ldst), "v"(p.A + ge) : "memory");
        if (dual) {
          unsigned ldst2 = (unsigned)(unsigned long long)(As2 + (m*AP + k));
          asm volatile("global_load_async_to_lds_b128 %0, %1, off"
                       :: "v"(ldst2), "v"(p.A2 + ge) : "memory");
        }
      }
    } else {
      // ---- scalar bounds-checked staging ---------------------------------
      #pragma unroll
      for (int i = 0; i < 16; ++i) {
        int idx = tid + i*256;
        int m = idx >> 5, k = idx & 31;
        bool ok = (m0+m < p.M) && (k0+k < p.K);
        long ga = offA + (long)(m0+m)*p.sAm + (long)(k0+k)*p.sAk;
        As[m*AP+k] = ok ? p.A[ga] : (__bf16)0.f;
        if (dual) As2[m*AP+k] = ok ? p.A2[ga] : (__bf16)0.f;
      }
    }
    // ---- B tile staged TRANSPOSED: Bs[n][k] so fragments are contiguous --
    #pragma unroll
    for (int i = 0; i < 8; ++i) {
      int idx = tid + i*256;
      int k = idx >> 6, n = idx & 63;
      bool ok = (k0+k < p.K) && (n0+n < p.N);
      long gb = offB + (long)(k0+k)*p.sBk + (long)(n0+n)*p.sBn;
      Bs[n*KP+k] = ok ? p.B[gb] : (__bf16)0.f;
      if (dual) Bs2[n*KP+k] = ok ? p.B2[gb] : (__bf16)0.f;
    }
    // hint next K tile of A into cache (global_prefetch_b8)
    if (k0 + KT < p.K) {
      long ga = offA + (long)(m0 + (tid>>5))*p.sAm + (long)(k0+KT)*p.sAk;
      __builtin_prefetch(p.A + ga, 0, 0);
    }
    if (fastA) asm volatile("s_wait_asynccnt 0x0" ::: "memory");
    __syncthreads();

    // ---- build WMMA fragments in contiguous k order (same for A and B) --------------
    // A 16x32: lanes 0-15 row=r, elem i -> k = hi*16 + i  (contiguous in As[m][k])
    // B 32x16: lane col = r, elem i -> k = hi*16 + i  (contiguous in Bs[n][k])
    v16bf fa[2], fb[2];
    #pragma unroll
    for (int mi = 0; mi < 2; ++mi) {
      int row = wm*32 + mi*16 + r;
      #pragma unroll
      for (int i = 0; i < 8; ++i) {
        fa[mi][i]   = As[row*AP + hi*16 + i];
        fa[mi][i+8] = As[row*AP + hi*16 + 8 + i];
      }
    }
    #pragma unroll
    for (int ni = 0; ni < 2; ++ni) {
      int col = wn*32 + ni*16 + r;
      #pragma unroll
      for (int i = 0; i < 16; ++i)
        fb[ni][i] = Bs[col*KP + hi*16 + i];
    }
    #pragma unroll
    for (int mi = 0; mi < 2; ++mi)
      #pragma unroll
      for (int ni = 0; ni < 2; ++ni)
        acc[mi][ni] = __builtin_amdgcn_wmma_f32_16x16x32_bf16(
            false, fa[mi], false, fb[ni], (short)0, acc[mi][ni], false, false);

    if (dual) {
      v16bf fa2[2], fb2[2];
      #pragma unroll
      for (int mi = 0; mi < 2; ++mi) {
        int row = wm*32 + mi*16 + r;
        #pragma unroll
        for (int i = 0; i < 8; ++i) {
          fa2[mi][i]   = As2[row*AP + hi*16 + i];
          fa2[mi][i+8] = As2[row*AP + hi*16 + 8 + i];
        }
      }
      #pragma unroll
      for (int ni = 0; ni < 2; ++ni) {
        int col = wn*32 + ni*16 + r;
        #pragma unroll
        for (int i = 0; i < 16; ++i)
          fb2[ni][i] = Bs2[col*KP + hi*16 + i];
      }
      #pragma unroll
      for (int mi = 0; mi < 2; ++mi)
        #pragma unroll
        for (int ni = 0; ni < 2; ++ni)
          acc[mi][ni] = __builtin_amdgcn_wmma_f32_16x16x32_bf16(
              false, fa2[mi], false, fb2[ni], (short)0, acc[mi][ni], false, false);
    }
    __syncthreads();
  }

  // ---- epilogue: bias -> StarReLU -> *mult -> +residual -> store ----------
  float ss = 0.f, sb = 0.f;
  if (p.use_star) { ss = *p.star_s; sb = *p.star_b; }
  #pragma unroll
  for (int mi = 0; mi < 2; ++mi)
    #pragma unroll
    for (int ni = 0; ni < 2; ++ni)
      #pragma unroll
      for (int v = 0; v < 8; ++v) {
        int m = m0 + wm*32 + mi*16 + hi*8 + v;   // C/D: M = vgpr + 8*hi
        int n = n0 + wn*32 + ni*16 + r;          // C/D: N = lane&15
        if (m < p.M && n < p.N) {
          float val = acc[mi][ni][v];
          if (p.bias) val += p.bias[(long)z*p.sBias + n];
          if (p.use_star) { float t = fmaxf(val, 0.f); val = ss*t*t + sb; }
          long gc = offC + (long)m*p.sCm + (long)n*p.sCn;
          if (p.mult)   val *= (float)p.mult[gc];
          if (p.addres) val += p.addres[gc];
          if (p.Cf) p.Cf[gc] = val;
          if (p.Cb) p.Cb[gc] = (__bf16)val;
        }
      }
}

// ---- LayerNorm over last dim (256): one wave32 per token -------------------
__global__ __launch_bounds__(256) void kln(const float* __restrict__ x,
                                           const float* __restrict__ g,
                                           const float* __restrict__ b,
                                           __bf16* __restrict__ y, int T) {
  int wave = threadIdx.x >> 5, lane = threadIdx.x & 31;
  int t = blockIdx.x * 8 + wave;
  if (t >= T) return;
  const float* xp = x + (long)t * 256;
  float v[8], s = 0.f, s2 = 0.f;
  #pragma unroll
  for (int i = 0; i < 8; ++i) { v[i] = xp[lane + i*32]; s += v[i]; s2 += v[i]*v[i]; }
  #pragma unroll
  for (int m = 16; m >= 1; m >>= 1) { s += __shfl_xor(s, m, 32); s2 += __shfl_xor(s2, m, 32); }
  float mu  = s  * (1.f/256.f);
  float var = s2 * (1.f/256.f) - mu*mu;
  float rs  = rsqrtf(var + 1e-5f);
  __bf16* yp = y + (long)t * 256;
  #pragma unroll
  for (int i = 0; i < 8; ++i) {
    int c = lane + i*32;
    yp[c] = (__bf16)((v[i]-mu)*rs*g[c] + b[c]);
  }
}

// ---- f32 -> bf16 weight conversion ----------------------------------------
__global__ void kf2b(const float* __restrict__ s, __bf16* __restrict__ d, int n) {
  int i = blockIdx.x * 256 + threadIdx.x;
  if (i < n) d[i] = (__bf16)s[i];
}

// ---- twiddle tables, ortho scale 0.1 per axis folded in -------------------
// layout (bf16 elems): T1=cos[100x100] @0, T2=sin @10000, T3=-sin @20000,
// T4=cos @30000, TROW[100x128] @40000 (cols 0..50 cos, 64..114 -sin),
// TIRA[51x100] @52800 (a_k*cos), TIRB @57900 (-a_k*sin), a_0=a_50=1 else 2.
__global__ void ktwiddle(__bf16* t) {
  int i = blockIdx.x * 256 + threadIdx.x;
  const float C = 2.f * 3.14159265358979323846f / 100.f;
  if (i < 40000) {
    int tab = i / 10000, j = i % 10000;
    int h = j / 100, k = j % 100;
    float ang = C * (float)(h * k);
    float c = cosf(ang) * 0.1f, s = sinf(ang) * 0.1f;
    float v = (tab == 0) ? c : (tab == 1) ? s : (tab == 2) ? -s : c;
    t[i] = (__bf16)v;
  } else if (i < 52800) {
    int j = i - 40000; int w = j / 128, n = j % 128;
    float v = 0.f;
    if (n < 51)              v =  cosf(C * (float)(w * n)) * 0.1f;
    else if (n >= 64 && n < 115) v = -sinf(C * (float)(w * (n-64))) * 0.1f;
    t[i] = (__bf16)v;
  } else if (i < 63000) {
    int j = i - 52800; int tab = j / 5100; int q = j % 5100;
    int kw = q / 100, w = q % 100;
    float a = (kw == 0 || kw == 50) ? 1.f : 2.f;
    float ang = C * (float)(w * kw);
    t[i] = (__bf16)(tab == 0 ? a * cosf(ang) * 0.1f : -a * sinf(ang) * 0.1f);
  }
}

static void run_gemm(const GemmP& p, int nbatch, hipStream_t s) {
  dim3 grid((p.M + MT - 1) / MT, (p.N + NT - 1) / NT, nbatch);
  kgemm<<<grid, dim3(256), 0, s>>>(p);
}

extern "C" void kernel_launch(void* const* d_in, const int* in_sizes, int n_in,
                              void* d_out, int out_size, void* d_ws, size_t ws_size,
                              hipStream_t stream) {
  (void)in_sizes; (void)n_in; (void)out_size; (void)ws_size;
  const float* x     = (const float*)d_in[0];
  const float* ln1_g = (const float*)d_in[1];
  const float* ln1_b = (const float*)d_in[2];
  const float* pw1_w = (const float*)d_in[3];
  const float* act1_s= (const float*)d_in[4];
  const float* act1_b= (const float*)d_in[5];
  const float* w1    = (const float*)d_in[6];
  const float* w11   = (const float*)d_in[7];
  const float* b1    = (const float*)d_in[8];
  const float* w2    = (const float*)d_in[9];
  const float* w22   = (const float*)d_in[10];
  const float* b2    = (const float*)d_in[11];
  const float* actr_s= (const float*)d_in[12];
  const float* actr_b= (const float*)d_in[13];
  const float* acti_s= (const float*)d_in[14];
  const float* acti_b= (const float*)d_in[15];
  const float* pw2_w = (const float*)d_in[16];
  const float* ln2_g = (const float*)d_in[17];
  const float* ln2_b = (const float*)d_in[18];
  const float* fc1_w = (const float*)d_in[19];
  const float* mlp_s = (const float*)d_in[20];
  const float* mlp_b = (const float*)d_in[21];
  const float* fc2_w = (const float*)d_in[22];

  constexpr int  T  = 80000;   // 8*100*100 tokens
  constexpr int  BH = 800;     // batch*rows
  constexpr int  FP = 40800;   // 8*100*51 frequency points

  // ---- workspace layout (bytes) -------------------------------------------
  char* w = (char*)d_ws;
  size_t o = 0;
  __bf16* pw1bf = (__bf16*)(w + o); o += 256*512*2;
  __bf16* pw2bf = (__bf16*)(w + o); o += 512*256*2;
  __bf16* fc1bf = (__bf16*)(w + o); o += 256*1024*2;
  __bf16* fc2bf = (__bf16*)(w + o); o += 1024*256*2;
  __bf16* w1bf  = (__bf16*)(w + o); o += 65536*2;
  __bf16* w11bf = (__bf16*)(w + o); o += 65536*2;
  __bf16* w2bf  = (__bf16*)(w + o); o += 65536*2;
  __bf16* w22bf = (__bf16*)(w + o); o += 65536*2;
  __bf16* twd   = (__bf16*)(w + o); o += 131072;        // 63000 elems, padded
  __bf16* ybf   = (__bf16*)(w + o); o += (size_t)T*256*2;      // LN1 out; reused for LN2 out
  float*  h32   = (float* )(w + o);                            // pw1 out f32 (skip)
  __bf16* m1bf  = (__bf16*)(w + o); o += (size_t)T*512*4;      // m1 (bf16) reuses h32 region later
  __bf16* hbf   = (__bf16*)(w + o); o += (size_t)T*512*2;      // pw1 out bf16
  __bf16* F1    = (__bf16*)(w + o);                            // row-FFT out [800][512][128]
  __bf16* Grbf  = (__bf16*)(w + o);                            // iFFT col out reuses F1 region
  __bf16* Gibf  = (__bf16*)(w + o + (size_t)FP*512*2);
  o += 800ull*512*128*2;
  __bf16* frbf  = (__bf16*)(w + o); o += (size_t)FP*512*2;
  __bf16* fibf  = (__bf16*)(w + o); o += (size_t)FP*512*2;
  __bf16* o1r   = (__bf16*)(w + o); o += (size_t)FP*512*2;
  __bf16* o1i   = (__bf16*)(w + o); o += (size_t)FP*512*2;
  __bf16* o2r   = (__bf16*)(w + o); o += (size_t)FP*512*2;
  __bf16* o2i   = (__bf16*)(w + o); o += (size_t)FP*512*2;
  __bf16* hsum  = (__bf16*)(w + o); o += (size_t)T*512*2;      // irfft + skip (bf16)
  float*  x1    = (float* )(w + o); o += (size_t)T*256*4;      // x + mix (f32)

  __bf16* T1 = twd, *T2 = twd+10000, *T3 = twd+20000, *T4 = twd+30000;
  __bf16* TROW = twd+40000, *TIRA = twd+52800, *TIRB = twd+57900;

  // ---- prep: twiddles + bf16 weights --------------------------------------
  ktwiddle<<<dim3(247), dim3(256), 0, stream>>>(twd);
  kf2b<<<dim3(512),  dim3(256), 0, stream>>>(pw1_w, pw1bf, 131072);
  kf2b<<<dim3(512),  dim3(256), 0, stream>>>(pw2_w, pw2bf, 131072);
  kf2b<<<dim3(1024), dim3(256), 0, stream>>>(fc1_w, fc1bf, 262144);
  kf2b<<<dim3(1024), dim3(256), 0, stream>>>(fc2_w, fc2bf, 262144);
  kf2b<<<dim3(256),  dim3(256), 0, stream>>>(w1,  w1bf,  65536);
  kf2b<<<dim3(256),  dim3(256), 0, stream>>>(w11, w11bf, 65536);
  kf2b<<<dim3(256),  dim3(256), 0, stream>>>(w2,  w2bf,  65536);
  kf2b<<<dim3(256),  dim3(256), 0, stream>>>(w22, w22bf, 65536);

  // ---- LN1 ----------------------------------------------------------------
  kln<<<dim3(10000), dim3(256), 0, stream>>>(x, ln1_g, ln1_b, ybf, T);

  GemmP p;

  // ---- pw1 + StarReLU: h = star(y @ pw1_w)  [80000x256x512] ---------------
  p = GemmP{};
  p.A = ybf; p.sAm = 256; p.sAk = 1; p.batchDiv = 1<<30;
  p.B = pw1bf; p.sBk = 512; p.sBn = 1;
  p.Cf = h32; p.Cb = hbf; p.sCm = 512; p.sCn = 1;
  p.star_s = act1_s; p.star_b = act1_b; p.use_star = 1;
  p.M = T; p.N = 512; p.K = 256;
  run_gemm(p, 1, stream);

  // ---- row rFFT as GEMM: per (b,h): [c=512 x w=100] @ TROW[100x128] -------
  p = GemmP{};
  p.A = hbf; p.sAm = 1; p.sAk = 512; p.sAb1 = 51200; p.batchDiv = 1<<30;
  p.B = TROW; p.sBk = 128; p.sBn = 1;
  p.Cb = F1; p.sCm = 128; p.sCn = 1; p.sCb1 = 65536;
  p.M = 512; p.N = 128; p.K = 100;
  run_gemm(p, BH, stream);

  // ---- column FFT (complex): per (b,kw): Fr = F1r*cos + F1i*sin -----------
  p = GemmP{};
  p.A = F1; p.A2 = F1 + 64;
  p.sAm = 128; p.sAk = 65536; p.sAb1 = 1; p.sAb2 = 6553600; p.batchDiv = 51;
  p.B = T1; p.B2 = T2; p.sBk = 100; p.sBn = 1;
  p.Cb = frbf; p.sCm = 1; p.sCn = 26112; p.sCb1 = 512; p.sCb2 = 2611200;
  p.M = 512; p.N = 100; p.K = 100;
  run_gemm(p, 408, stream);
  // Fi = -F1r*sin + F1i*cos
  p.B = T3; p.B2 = T4; p.Cb = fibf;
  run_gemm(p, 408, stream);

  // ---- ASMM stage 1: o1r = star(fr@w1[0] + fi@w1[1] + b1[0]) per block ----
  p = GemmP{};
  p.A = frbf; p.A2 = fibf; p.sAm = 512; p.sAk = 1; p.sAb1 = 64; p.batchDiv = 1<<30;
  p.B = w1bf; p.B2 = w1bf + 32768; p.sBk = 64; p.sBn = 1; p.sBb1 = 4096;
  p.bias = b1; p.sBias = 64;
  p.star_s = actr_s; p.star_b = actr_b; p.use_star = 1;
  p.Cb = o1r; p.sCm = 512; p.sCn = 1; p.sCb1 = 64;
  p.M = FP; p.N = 64; p.K = 64;
  run_gemm(p, 8, stream);
  // o1i
  p.A = fibf; p.A2 = frbf;
  p.B = w11bf; p.B2 = w11bf + 32768;
  p.bias = b1 + 512;
  p.star_s = acti_s; p.star_b = acti_b;
  p.Cb = o1i;
  run_gemm(p, 8, stream);

  // ---- ASMM stage 2: o2r = (o1r@w2[0] + o1i@w2[1] + b2[0]) * fr -----------
  p = GemmP{};
  p.A = o1r; p.A2 = o1i; p.sAm = 512; p.sAk = 1; p.sAb1 = 64; p.batchDiv = 1<<30;
  p.B = w2bf; p.B2 = w2bf + 32768; p.sBk = 64; p.sBn = 1; p.sBb1 = 4096;
  p.bias = b2; p.sBias = 64;
  p.mult = frbf;
  p.Cb = o2r; p.sCm = 512; p.sCn = 1; p.sCb1 = 64;
  p.M = FP; p.N = 64; p.K = 64;
  run_gemm(p, 8, stream);
  // o2i
  p.A = o1i; p.A2 = o1r;
  p.B = w22bf; p.B2 = w22bf + 32768;
  p.bias = b2 + 512;
  p.mult = fibf;
  p.Cb = o2i;
  run_gemm(p, 8, stream);

  // ---- inverse column FFT: Gr = o2r*cos - o2i*sin -------------------------
  p = GemmP{};
  p.A = o2r; p.A2 = o2i;
  p.sAm = 1; p.sAk = 26112; p.sAb1 = 512; p.sAb2 = 2611200; p.batchDiv = 51;
  p.B = T1; p.B2 = T3; p.sBk = 100; p.sBn = 1;
  p.Cb = Grbf; p.sCm = 1; p.sCn = 26112; p.sCb1 = 512; p.sCb2 = 2611200;
  p.M = 512; p.N = 100; p.K = 100;
  run_gemm(p, 408, stream);
  // Gi = o2r*sin + o2i*cos
  p.B = T2; p.B2 = T4; p.Cb = Gibf;
  run_gemm(p, 408, stream);

  // ---- row irFFT (hermitian-folded) + skip: hsum = irfft + h --------------
  p = GemmP{};
  p.A = Grbf; p.A2 = Gibf; p.sAm = 1; p.sAk = 512; p.sAb1 = 26112; p.batchDiv = 1<<30;
  p.B = TIRA; p.B2 = TIRB; p.sBk = 100; p.sBn = 1;
  p.addres = h32;
  p.Cb = hsum; p.sCm = 1; p.sCn = 512; p.sCb1 = 51200;
  p.M = 512; p.N = 100; p.K = 51;
  run_gemm(p, BH, stream);

  // ---- pw2 + residual: x1 = x + hsum @ pw2_w ------------------------------
  p = GemmP{};
  p.A = hsum; p.sAm = 512; p.sAk = 1; p.batchDiv = 1<<30;
  p.B = pw2bf; p.sBk = 256; p.sBn = 1;
  p.addres = x;
  p.Cf = x1; p.sCm = 256; p.sCn = 1;
  p.M = T; p.N = 256; p.K = 512;
  run_gemm(p, 1, stream);

  // ---- LN2 (reuse ybf) ----------------------------------------------------
  kln<<<dim3(10000), dim3(256), 0, stream>>>(x1, ln2_g, ln2_b, ybf, T);

  // ---- fc1 + StarReLU -----------------------------------------------------
  p = GemmP{};
  p.A = ybf; p.sAm = 256; p.sAk = 1; p.batchDiv = 1<<30;
  p.B = fc1bf; p.sBk = 1024; p.sBn = 1;
  p.star_s = mlp_s; p.star_b = mlp_b; p.use_star = 1;
  p.Cb = m1bf; p.sCm = 1024; p.sCn = 1;
  p.M = T; p.N = 1024; p.K = 256;
  run_gemm(p, 1, stream);

  // ---- fc2 + residual -> d_out --------------------------------------------
  p = GemmP{};
  p.A = m1bf; p.sAm = 1024; p.sAk = 1; p.batchDiv = 1<<30;
  p.B = fc2bf; p.sBk = 256; p.sBn = 1;
  p.addres = x1;
  p.Cf = (float*)d_out; p.sCm = 256; p.sCn = 1;
  p.M = T; p.N = 256; p.K = 1024;
  run_gemm(p, 1, stream);
}
